// Adapter_20701742366997
// MI455X (gfx1250) — matrix operand order repaired and checked by reading, not compiled, not run on hardware
//
#include <hip/hip_runtime.h>
#include <hip/hip_bf16.h>
#include <hip/hip_fp16.h>

typedef __attribute__((ext_vector_type(16))) _Float16 v16h;
typedef __attribute__((ext_vector_type(8)))  _Float16 v8h;
typedef __attribute__((ext_vector_type(8)))  float    v8f;
typedef __attribute__((ext_vector_type(4)))  float    v4f;

#define DZ      1024
#define DM      4096
#define NROWS   16384
#define LN_EPS  1e-5f
#define CLAMP_V 3.0f

// ---------------------------------------------------------------------------
// Kernel 1: LayerNorm each row of z (NROWS x DZ), split result into f16 hi/lo
// ---------------------------------------------------------------------------
__global__ __launch_bounds__(256) void ln_split_kernel(
    const float* __restrict__ z, const float* __restrict__ gamma,
    const float* __restrict__ beta,
    _Float16* __restrict__ zh, _Float16* __restrict__ zl)
{
    const int row  = blockIdx.x;
    const int t    = threadIdx.x;
    const int lane = t & 31;
    const int wave = t >> 5;

    const float* zr = z + (size_t)row * DZ;
    v4f v = *(const v4f*)(zr + t * 4);

    float s  = v[0] + v[1] + v[2] + v[3];
    float s2 = v[0]*v[0] + v[1]*v[1] + v[2]*v[2] + v[3]*v[3];
    #pragma unroll
    for (int off = 16; off > 0; off >>= 1) {
        s  += __shfl_xor(s,  off, 32);
        s2 += __shfl_xor(s2, off, 32);
    }

    __shared__ float ls[8], ls2[8];
    if (lane == 0) { ls[wave] = s; ls2[wave] = s2; }
    __syncthreads();

    float tot = 0.f, tot2 = 0.f;
    #pragma unroll
    for (int i = 0; i < 8; i++) { tot += ls[i]; tot2 += ls2[i]; }

    const float mu   = tot * (1.0f / DZ);
    const float var  = tot2 * (1.0f / DZ) - mu * mu;
    const float rstd = rsqrtf(var + LN_EPS);

    #pragma unroll
    for (int j = 0; j < 4; j++) {
        const int   k  = t * 4 + j;
        const float zn = (v[j] - mu) * rstd * gamma[k] + beta[k];
        const _Float16 h = (_Float16)zn;
        const _Float16 l = (_Float16)(zn - (float)h);
        zh[(size_t)row * DZ + k] = h;
        zl[(size_t)row * DZ + k] = l;
    }
}

// ---------------------------------------------------------------------------
// Kernel 2: transpose W (DZ x DM, row-major) -> Wt (DM x DZ) as f16 hi/lo
// ---------------------------------------------------------------------------
__global__ __launch_bounds__(256) void wt_split_kernel(
    const float* __restrict__ W,
    _Float16* __restrict__ wh, _Float16* __restrict__ wl)
{
    __shared__ float tile[32][33];
    const int tx = threadIdx.x;     // 0..31
    const int ty = threadIdx.y;     // 0..7
    const int n0 = blockIdx.x * 32; // N tile
    const int k0 = blockIdx.y * 32; // K tile

    #pragma unroll
    for (int j = 0; j < 32; j += 8)
        tile[ty + j][tx] = W[(size_t)(k0 + ty + j) * DM + n0 + tx];
    __syncthreads();

    #pragma unroll
    for (int j = 0; j < 32; j += 8) {
        const float val = tile[tx][ty + j];       // [k-local][n-local]
        const _Float16 h = (_Float16)val;
        const _Float16 l = (_Float16)(val - (float)h);
        const size_t o = (size_t)(n0 + ty + j) * DZ + (k0 + tx);
        wh[o] = h;
        wl[o] = l;
    }
}

// ---------------------------------------------------------------------------
// Kernel 3: GEMM via v_wmma_f32_16x16x32_f16 with hi/lo split (3 WMMA / tile)
// Block = 256 threads (8 waves). Block tile 128(M) x 64(N); wave tile 16 x 64.
// ---------------------------------------------------------------------------
__global__ __launch_bounds__(256) void gemm_wmma_kernel(
    const _Float16* __restrict__ zh, const _Float16* __restrict__ zl,
    const _Float16* __restrict__ wh, const _Float16* __restrict__ wl,
    const float* __restrict__ bias, const float* __restrict__ scale_p,
    float* __restrict__ out)
{
    const int lane   = threadIdx.x & 31;
    const int wave   = threadIdx.x >> 5;
    const int m_base = blockIdx.y * 128 + wave * 16;
    const int n_base = blockIdx.x * 64;

    const int lmod = lane & 15;
    const int lhi  = lane >> 4;

    // A-fragment (16x32, this code's contiguous K order, same as B): lanes 0-15 hold row M=lane,
    // K = [lhi*16 .. +16) per 32-wide K step.
    const _Float16* arow_h = zh + (size_t)(m_base + lmod) * DZ + lhi * 16;
    const _Float16* arow_l = zl + (size_t)(m_base + lmod) * DZ + lhi * 16;

    v8f acc[4] = {};

    for (int k = 0; k < DZ; k += 32) {
        v16h ah, al;
        *(v8h*)&ah       = *(const v8h*)(arow_h + k);
        *((v8h*)&ah + 1) = *(const v8h*)(arow_h + k + 8);
        *(v8h*)&al       = *(const v8h*)(arow_l + k);
        *((v8h*)&al + 1) = *(const v8h*)(arow_l + k + 8);

        #pragma unroll
        for (int nt = 0; nt < 4; nt++) {
            // B-fragment (32x16): lane holds column n = n_base+nt*16+lmod,
            // 16 contiguous K values starting at k + lhi*16.
            const size_t bo = (size_t)(n_base + nt * 16 + lmod) * DZ + k + lhi * 16;
            const v16h bh = *(const v16h*)(wh + bo);
            const v16h bl = *(const v16h*)(wl + bo);

            acc[nt] = __builtin_amdgcn_wmma_f32_16x16x32_f16(
                false, ah, false, bh, (short)0, acc[nt], false, false);
            acc[nt] = __builtin_amdgcn_wmma_f32_16x16x32_f16(
                false, ah, false, bl, (short)0, acc[nt], false, false);
            acc[nt] = __builtin_amdgcn_wmma_f32_16x16x32_f16(
                false, al, false, bh, (short)0, acc[nt], false, false);
        }
    }

    // Epilogue: C/D layout -> lane = N column, vgpr index v = M row
    // (lanes 0-15: M = m_base+v ; lanes 16-31: M = m_base+8+v)
    const float scale = scale_p[0];
    const int   row0  = m_base + lhi * 8;
    #pragma unroll
    for (int nt = 0; nt < 4; nt++) {
        const int   col = n_base + nt * 16 + lmod;
        const float bv  = bias[col];
        #pragma unroll
        for (int v = 0; v < 8; v++) {
            const float x = (acc[nt][v] + bv) * scale;
            out[(size_t)(row0 + v) * DM + col] =
                tanhf(x * (1.0f / CLAMP_V)) * CLAMP_V;
        }
    }
}

// ---------------------------------------------------------------------------
extern "C" void kernel_launch(void* const* d_in, const int* in_sizes, int n_in,
                              void* d_out, int out_size, void* d_ws, size_t ws_size,
                              hipStream_t stream) {
    const float* z     = (const float*)d_in[0];
    const float* gamma = (const float*)d_in[1];
    const float* beta  = (const float*)d_in[2];
    const float* W     = (const float*)d_in[3];
    const float* b     = (const float*)d_in[4];
    const float* scale = (const float*)d_in[5];
    float*       out   = (float*)d_out;

    char* ws = (char*)d_ws;
    const size_t zbytes = (size_t)NROWS * DZ * sizeof(_Float16); // 32 MB each
    const size_t wbytes = (size_t)DM * DZ * sizeof(_Float16);    //  8 MB each
    _Float16* zh = (_Float16*)(ws);
    _Float16* zl = (_Float16*)(ws + zbytes);
    _Float16* wh = (_Float16*)(ws + 2 * zbytes);
    _Float16* wl = (_Float16*)(ws + 2 * zbytes + wbytes);

    ln_split_kernel<<<NROWS, 256, 0, stream>>>(z, gamma, beta, zh, zl);
    wt_split_kernel<<<dim3(DM / 32, DZ / 32), dim3(32, 8), 0, stream>>>(W, wh, wl);
    gemm_wmma_kernel<<<dim3(DM / 64, NROWS / 128), 256, 0, stream>>>(
        zh, zl, wh, wl, b, scale, out);
}
